// MyGRU_36773509988472
// MI455X (gfx1250) — matrix operand order repaired and checked by reading, compiled, not run on hardware
//
#include <hip/hip_runtime.h>
#include <hip/hip_bf16.h>

// Problem dims (fixed by the reference)
#define SEQ    512
#define BATCH  128
#define INPUT  1024
#define HIDDEN 1024
#define GATES  (3 * HIDDEN)   // 3072

typedef __attribute__((ext_vector_type(16))) __bf16 v16bf;
typedef __attribute__((ext_vector_type(8)))  __bf16 v8bf;
typedef __attribute__((ext_vector_type(8)))  float  v8f;

// ---------------------------------------------------------------------------
// Fragment loaders (contiguous k order: lane half h holds k = 16h + s, wave32)
// ---------------------------------------------------------------------------

// A-matrix 16x32 bf16 from a row-major bf16 source.
// lane L<16: M=L,      K = {kBase+0..15}
// lane L>=16:M=L-16,   K = {kBase+16..31}
static __device__ __forceinline__
v16bf load_a_frag_bf16(const __bf16* __restrict__ A, int lda,
                       int mBase, int kBase, int lane) {
    const int m    = mBase + (lane & 15);
    const int kOff = ((lane >> 4) & 1) * 16;
    const __bf16* __restrict__ p = A + (size_t)m * lda + kBase + kOff;
    v8bf lo = *reinterpret_cast<const v8bf*>(p);
    v8bf hi = *reinterpret_cast<const v8bf*>(p + 8);
    return __builtin_shufflevector(lo, hi, 0, 1, 2, 3, 4, 5, 6, 7,
                                           8, 9, 10, 11, 12, 13, 14, 15);
}

// B-matrix 32x16 bf16 where B = W^T, W row-major [N x K] bf16.
static __device__ __forceinline__
v16bf load_b_frag_bf16(const __bf16* __restrict__ W, int ldw,
                       int nBase, int kBase, int lane) {
    const int n = nBase + (lane & 15);
    const int k = kBase + ((lane >> 4) & 1) * 16;
    return *reinterpret_cast<const v16bf*>(W + (size_t)n * ldw + k);
}

static __device__ __forceinline__ float sigmoidf(float x) {
    return 1.0f / (1.0f + __expf(-x));
}

// ---------------------------------------------------------------------------
// Kernel A: f32 -> bf16 conversion (weights, X, h0)
// ---------------------------------------------------------------------------
__global__ void cvt_f32_to_bf16(const float* __restrict__ src,
                                __bf16* __restrict__ dst, int n) {
    int i = blockIdx.x * blockDim.x + threadIdx.x;
    if (i < n) dst[i] = (__bf16)src[i];
}

// ---------------------------------------------------------------------------
// Kernel B: x_proj[S*B, 3H] = Xbf[S*B, I] @ W_ih^T + b_ih
// One wave computes a 16x64 output tile; reuse_a hint on trailing WMMAs.
// ---------------------------------------------------------------------------
#define GEMM_N4 (GATES / 64)  // 48 column groups of 64
__global__ void xproj_gemm(const __bf16* __restrict__ Xbf,
                           const __bf16* __restrict__ Wih,
                           const float* __restrict__ b_ih,
                           float* __restrict__ xproj) {
    const int lane = threadIdx.x & 31;
    const int wave = blockIdx.x * (blockDim.x >> 5) + (threadIdx.x >> 5);
    const int mT   = wave / GEMM_N4;       // 0 .. 4095
    const int n4   = wave % GEMM_N4;       // 0 .. 47
    const int mBase = mT * 16;
    const int nBase = n4 * 64;

    v8f acc0 = {}, acc1 = {}, acc2 = {}, acc3 = {};
#pragma unroll 4
    for (int kb = 0; kb < INPUT; kb += 32) {
        v16bf a  = load_a_frag_bf16(Xbf, INPUT, mBase, kb, lane);
        v16bf b0 = load_b_frag_bf16(Wih, INPUT, nBase +  0, kb, lane);
        v16bf b1 = load_b_frag_bf16(Wih, INPUT, nBase + 16, kb, lane);
        v16bf b2 = load_b_frag_bf16(Wih, INPUT, nBase + 32, kb, lane);
        v16bf b3 = load_b_frag_bf16(Wih, INPUT, nBase + 48, kb, lane);
        acc0 = __builtin_amdgcn_wmma_f32_16x16x32_bf16(false, a, false, b0, (short)0, acc0, false, false);
        acc1 = __builtin_amdgcn_wmma_f32_16x16x32_bf16(false, a, false, b1, (short)0, acc1, true,  false);
        acc2 = __builtin_amdgcn_wmma_f32_16x16x32_bf16(false, a, false, b2, (short)0, acc2, true,  false);
        acc3 = __builtin_amdgcn_wmma_f32_16x16x32_bf16(false, a, false, b3, (short)0, acc3, true,  false);
    }

    const int mRow0 = mBase + ((lane >> 4) & 1) * 8;
    const int nCol  = lane & 15;
    v8f accs[4] = {acc0, acc1, acc2, acc3};
#pragma unroll
    for (int j = 0; j < 4; ++j) {
        const int n = nBase + 16 * j + nCol;
        const float bias = b_ih[n];
#pragma unroll
        for (int e = 0; e < 8; ++e) {
            xproj[(size_t)(mRow0 + e) * GATES + n] = accs[j][e] + bias;
        }
    }
}

// ---------------------------------------------------------------------------
// Kernel C: one GRU time step.
//   - A operand: bf16 shadow of h (straight b128 loads, no cvt on critical path)
//   - f32 master h for the z*h carry (accuracy over 512 steps)
//   - W_hh panel staged into LDS with global_load_async_to_lds_b128,
//     triple buffered, one barrier per k-step, one panel always in flight.
// Grid: 64 blocks (one per hidden tile), 256 threads = 8 waves (batch tiles).
// ---------------------------------------------------------------------------
__global__ void __launch_bounds__(256)
gru_step(const __bf16* __restrict__ hbf_in,  // [B, H] bf16 shadow of h_prev
         const float* __restrict__ h_prev,   // [B, H] f32 master
         const float* __restrict__ xproj_t,  // [B, 3H]
         const __bf16* __restrict__ Whh,     // [3H, H] bf16
         const float* __restrict__ b_hh,     // [3H]
         float* __restrict__ h_out,          // [B, H] f32 master out
         __bf16* __restrict__ hbf_out) {     // [B, H] bf16 shadow out
    __shared__ __bf16 Bpanel[3][3][16][32];  // 9 KB, triple buffered

    const int tid  = threadIdx.x;
    const int lane = tid & 31;
    const int wave = tid >> 5;               // 0..7 -> batch tile
    const int nT   = blockIdx.x;             // 0..63 -> hidden tile
    const int mBase = wave * 16;
    const int nBase = nT * 16;

    // Staging map: threads 0..191 each move one 16-byte chunk per k-step.
    const int srow   = tid >> 2;             // 0..47 = gate*16 + n
    const int schunk = tid & 3;              // 16B chunk within a 64B row
    const int sg = srow >> 4;
    const int sn = srow & 15;
    const __bf16* gsrc = Whh + (size_t)(sg * HIDDEN + nBase + sn) * HIDDEN + schunk * 8;

    auto stage = [&](int buf, int kb) {
        if (tid < 192) {
            unsigned lds_off = (unsigned)(size_t)&Bpanel[buf][sg][sn][schunk * 8];
            const __bf16* g = gsrc + kb;
            asm volatile("global_load_async_to_lds_b128 %0, %1, off"
                         :: "v"(lds_off), "v"(g) : "memory");
        }
    };

    const int NK = HIDDEN / 32;              // 32 k-steps
    stage(0, 0);
    stage(1, 32);

    v8f accR = {}, accZ = {}, accN = {};
    const int khalf = ((lane >> 4) & 1) * 16;
    const int nlane = lane & 15;
    for (int i = 0; i < NK; ++i) {
        // In-order async completion: <=1 leaves only stage(i+1) in flight.
        if (i + 1 < NK) {
            asm volatile("s_wait_asynccnt 0x1" ::: "memory");
        } else {
            asm volatile("s_wait_asynccnt 0x0" ::: "memory");
        }
        __syncthreads();   // everyone's stage(i) visible in LDS
        const int buf = i % 3;
        v16bf br = *reinterpret_cast<const v16bf*>(&Bpanel[buf][0][nlane][khalf]);
        v16bf bz = *reinterpret_cast<const v16bf*>(&Bpanel[buf][1][nlane][khalf]);
        v16bf bn = *reinterpret_cast<const v16bf*>(&Bpanel[buf][2][nlane][khalf]);
        v16bf a  = load_a_frag_bf16(hbf_in, HIDDEN, mBase, i * 32, lane);
        // Complete our LDS reads before signaling the next barrier (so a
        // later stage() into buf (i-1)%3 cannot race our reads).
        asm volatile("s_wait_dscnt 0x0" ::: "memory");
        if (i + 2 < NK) stage((i + 2) % 3, (i + 2) * 32);
        accR = __builtin_amdgcn_wmma_f32_16x16x32_bf16(false, a, false, br, (short)0, accR, false, false);
        accZ = __builtin_amdgcn_wmma_f32_16x16x32_bf16(false, a, false, bz, (short)0, accZ, true,  false);
        accN = __builtin_amdgcn_wmma_f32_16x16x32_bf16(false, a, false, bn, (short)0, accN, true,  false);
    }

    // Epilogue: fused GRU gates (all f32), dual store (f32 master + bf16 shadow).
    const int j     = nBase + nlane;                  // hidden index
    const int mRow0 = mBase + ((lane >> 4) & 1) * 8;  // first batch row for e=0
    const float bR = b_hh[0 * HIDDEN + j];
    const float bZ = b_hh[1 * HIDDEN + j];
    const float bN = b_hh[2 * HIDDEN + j];
#pragma unroll
    for (int e = 0; e < 8; ++e) {
        const int m = mRow0 + e;
        const float xr = xproj_t[(size_t)m * GATES + 0 * HIDDEN + j];
        const float xz = xproj_t[(size_t)m * GATES + 1 * HIDDEN + j];
        const float xn = xproj_t[(size_t)m * GATES + 2 * HIDDEN + j];
        const float r  = sigmoidf(xr + accR[e] + bR);
        const float z  = sigmoidf(xz + accZ[e] + bZ);
        const float n  = tanhf(xn + r * (accN[e] + bN));
        const float hp = h_prev[(size_t)m * HIDDEN + j];
        const float hn = (1.0f - z) * n + z * hp;
        h_out[(size_t)m * HIDDEN + j]   = hn;
        hbf_out[(size_t)m * HIDDEN + j] = (__bf16)hn;
    }
}

// ---------------------------------------------------------------------------
// Host-side launcher
// ---------------------------------------------------------------------------
extern "C" void kernel_launch(void* const* d_in, const int* in_sizes, int n_in,
                              void* d_out, int out_size, void* d_ws, size_t ws_size,
                              hipStream_t stream) {
    const float* X    = (const float*)d_in[0];  // [S, B, I]
    const float* h0   = (const float*)d_in[1];  // [1, B, H]
    const float* W_ih = (const float*)d_in[2];  // [3H, I]
    const float* W_hh = (const float*)d_in[3];  // [3H, H]
    const float* b_ih = (const float*)d_in[4];  // [3H]
    const float* b_hh = (const float*)d_in[5];  // [3H]
    float* out = (float*)d_out;                 // [S*B*H] output ++ [B*H] h_last

    // Workspace layout
    __bf16* wih_bf = (__bf16*)d_ws;                              // 6 MB
    __bf16* whh_bf = wih_bf + (size_t)GATES * INPUT;             // 6 MB
    __bf16* hbf0   = whh_bf + (size_t)GATES * HIDDEN;            // 256 KB
    __bf16* hbf1   = hbf0 + (size_t)BATCH * HIDDEN;              // 256 KB
    float*  xproj  = (float*)(hbf1 + (size_t)BATCH * HIDDEN);    // 805 MB
    __bf16* x_bf   = (__bf16*)(xproj + (size_t)SEQ * BATCH * GATES); // 134 MB

    // A) one-time conversions
    {
        int n = GATES * INPUT;
        cvt_f32_to_bf16<<<(n + 255) / 256, 256, 0, stream>>>(W_ih, wih_bf, n);
        n = GATES * HIDDEN;
        cvt_f32_to_bf16<<<(n + 255) / 256, 256, 0, stream>>>(W_hh, whh_bf, n);
        n = BATCH * HIDDEN;
        cvt_f32_to_bf16<<<(n + 255) / 256, 256, 0, stream>>>(h0, hbf0, n);
        size_t nx = (size_t)SEQ * BATCH * INPUT;
        cvt_f32_to_bf16<<<(int)((nx + 255) / 256), 256, 0, stream>>>(X, x_bf, (int)nx);
    }

    // B) big input-projection GEMM: x_proj = X @ W_ih^T + b_ih
    {
        const int mTiles = (SEQ * BATCH) / 16;     // 4096
        const int totalWaves = mTiles * GEMM_N4;   // 4096 * 48
        const int wavesPerBlock = 8;               // 256 threads
        const int blocks = totalWaves / wavesPerBlock;
        xproj_gemm<<<blocks, wavesPerBlock * 32, 0, stream>>>(x_bf, wih_bf, b_ih, xproj);
    }

    // C) sequential scan; f32 h chains through the output buffer, bf16 shadow
    //    ping-pongs between hbf0/hbf1.
    {
        const int blocks = HIDDEN / 16;            // 64 (one per hidden tile)
        __bf16* hbf[2] = {hbf0, hbf1};
        for (int t = 0; t < SEQ; ++t) {
            const float* hprev = (t == 0) ? h0 : out + (size_t)(t - 1) * BATCH * HIDDEN;
            gru_step<<<blocks, 256, 0, stream>>>(
                hbf[t & 1], hprev,
                xproj + (size_t)t * BATCH * GATES,
                whh_bf, b_hh,
                out + (size_t)t * BATCH * HIDDEN,
                hbf[(t + 1) & 1]);
        }
    }

    // h_last = output[S-1]
    hipMemcpyAsync(out + (size_t)SEQ * BATCH * HIDDEN,
                   out + (size_t)(SEQ - 1) * BATCH * HIDDEN,
                   (size_t)BATCH * HIDDEN * sizeof(float),
                   hipMemcpyDeviceToDevice, stream);
}
